// Temporal_Attention_layer_10359461118369
// MI455X (gfx1250) — matrix operand order repaired and checked by reading, not compiled, not run on hardware
//
#include <hip/hip_runtime.h>
#include <hip/hip_bf16.h>

typedef __attribute__((ext_vector_type(16))) __bf16 v16bf;
typedef __attribute__((ext_vector_type(8)))  float  v8f;

#define Bdim 32
#define Ndim 512
#define Fdim 16
#define Tdim 64
#define Sdim 8192   // N*F
#define Gdim 256    // 4*T

__device__ __forceinline__ unsigned hash32(unsigned x) {
  x += 0x9E3779B9u * 42u;          // fold in the reference's key(42)
  x ^= x >> 16; x *= 0x7feb352du;
  x ^= x >> 15; x *= 0x846ca68bu;
  x ^= x >> 16;
  return x;
}
__device__ __forceinline__ float u01f(unsigned h) {
  return (float)(h >> 8) * 0x1p-24f + 0x1p-25f;   // in (0,1)
}
// Branch-free fast sigmoid / tanh: v_exp_f32 + v_rcp_f32, no IEEE div expansion.
__device__ __forceinline__ float sigmoid_fast(float v) {
  return __builtin_amdgcn_rcpf(1.f + __expf(-v));
}
__device__ __forceinline__ float tanh_fast(float v) {
  // tanh(v) = 1 - 2/(exp(2v)+1); exp->inf/0 saturates correctly to +-1
  return 1.f - 2.f * __builtin_amdgcn_rcpf(__expf(2.f * v) + 1.f);
}

// Workgroup barrier that only drains LDS (DScnt). Unlike __syncthreads(),
// this does NOT wait for outstanding global loads/stores, so the x-prefetch
// and the ws_rhs streaming stores stay in flight across the barrier.
// All in-loop producer->consumer traffic in phase 1 goes through LDS, so
// dscnt ordering + the barrier is sufficient (ISA 5/15.5: S_BARRIER_SIGNAL/WAIT).
__device__ __forceinline__ void lds_barrier() {
  asm volatile("s_wait_dscnt 0x0\n\t"
               "s_barrier_signal -1\n\t"
               "s_barrier_wait -1" ::: "memory");
}

// Build a 16x32 bf16 A-operand fragment from a row-major bf16 LDS buffer
// (16 rows x 128 cols). lane<16 holds row m, K {kc*32+0..15};
// lane>=16 holds row m, K {kc*32+16..31} (matches contiguous B fill).
__device__ __forceinline__ v16bf load_a_frag(const __bf16* Abf, int m, int kc, int half) {
  const __bf16* Ab = Abf + m * 128 + kc * 32 + half * 16;
  v16bf a;
#pragma unroll
  for (int i = 0; i < 8; ++i) { a[i] = Ab[i]; a[8 + i] = Ab[8 + i]; }
  return a;
}

// ---------------------------------------------------------------------------
// Phase 1: persistent LSTM scan + p/gating + attention partial reductions.
// grid = 2 workgroups (16 batch rows each, M=16 for WMMA), 256 threads (8 waves).
// ---------------------------------------------------------------------------
__global__ __launch_bounds__(256, 1) void lstm_gate_kernel(
    const float* __restrict__ x,
    const float* __restrict__ U1,
    const float* __restrict__ U3,
    const float* __restrict__ W_ih,
    const float* __restrict__ W_hh,
    const float* __restrict__ b_ih,
    const float* __restrict__ b_hh,
    const float* __restrict__ W_lin,
    const float* __restrict__ b_lin,
    float* __restrict__ ws_rhs,    // (B,N,T)
    float* __restrict__ ws_lhsA)   // (B,T,F)
{
  __shared__ __bf16 Abf[16 * 128];       // [b][ x(0..63) | h(64..127) ]  4 KB
  __shared__ float  gatesS[16 * 256];    // 16 KB
  __shared__ float  pS[16 * 64];         // 4 KB
  __shared__ float  lAacc[16 * 1024];    // [f][b*64+t]  64 KB
  __shared__ float  U1s[512];
  __shared__ float  U3s[16];

  const int tid   = threadIdx.x;
  const int lane  = tid & 31;
  const int w     = tid >> 5;       // wave 0..7
  const int col   = lane & 15;
  const int half  = lane >> 4;
  const int bbase = blockIdx.x * 16;

  for (int i = tid; i < 512; i += 256) U1s[i] = U1[i];
  if (tid < 16) U3s[tid] = U3[tid];
  for (int i = tid; i < 16 * 128; i += 256) Abf[i] = (__bf16)0.f;
  for (int i = tid; i < 16 * 1024; i += 256) lAacc[i] = 0.f;

  // Gate-GEMM B operand (loop-invariant, held in VGPRs as bf16).
  // B column g = tile*16 + (lane&15); lane holds 16 consecutive K of that column,
  // K-half chosen by lane>>4 (contiguous k order, same as load_a_frag).
  v16bf Bg[2][4];
  float biasg[2];
#pragma unroll
  for (int tt = 0; tt < 2; ++tt) {
    const int g = (w * 2 + tt) * 16 + col;
    biasg[tt] = b_ih[g] + b_hh[g];
#pragma unroll
    for (int kc = 0; kc < 4; ++kc) {
      const int k0 = kc * 32 + half * 16;   // 16-aligned run: stays inside W_ih or W_hh
      const float* src = (k0 < 64) ? (W_ih + g * 64 + k0) : (W_hh + g * 64 + (k0 - 64));
      v16bf v;
#pragma unroll
      for (int i = 0; i < 16; ++i) v[i] = (__bf16)src[i];
      Bg[tt][kc] = v;
    }
  }
  // W_lin B operand for the p-GEMM (waves 0..3, one 16-col tile each).
  v16bf Bl[2];
  float biasl = 0.f;
  if (w < 4) {
    const int gp = w * 16 + col;
    biasl = b_lin[gp];
#pragma unroll
    for (int kc = 0; kc < 2; ++kc) {
      const int k0 = kc * 32 + half * 16;
      const float* src = W_lin + gp * 64 + k0;
      v16bf v;
#pragma unroll
      for (int i = 0; i < 16; ++i) v[i] = (__bf16)src[i];
      Bl[kc] = v;
    }
  }

  // Fixed per-thread element ownership: (bL, t0..t0+3)
  const int bL = tid >> 4;
  const int t0 = (tid & 15) * 4;
  const float* xrow = x + ((size_t)(bbase + bL) * Sdim) * Tdim + t0;
  float rhs_acc[4] = {0.f, 0.f, 0.f, 0.f};
  float creg[4]    = {0.f, 0.f, 0.f, 0.f};   // LSTM cell state lives in registers

  __syncthreads();

  float4 xr = *(const float4*)(xrow);   // prefetched x_s (s = 0)

  for (int s = 0; s < Sdim; ++s) {
    // ---- 1) stage x_s into A cols 0..63 (f32 -> bf16) ----
    Abf[bL * 128 + t0 + 0] = (__bf16)xr.x;
    Abf[bL * 128 + t0 + 1] = (__bf16)xr.y;
    Abf[bL * 128 + t0 + 2] = (__bf16)xr.z;
    Abf[bL * 128 + t0 + 3] = (__bf16)xr.w;

    // Prefetch next step's x: with the dscnt-only barrier this load stays in
    // flight across the whole step and is waited only at next iteration's use.
    const int snext = (s + 1 < Sdim) ? s + 1 : s;
    const float4 xr_next = *(const float4*)(xrow + (size_t)snext * Tdim);

    // Pre-compute Gumbel-softmax noise (independent of p) in the GEMM shadow.
    // G1-G0 of two iid Gumbels is a Logistic variate: log(u) - log(1-u).
    float Lns[4];
#pragma unroll
    for (int j = 0; j < 4; ++j) {
      const unsigned id = ((unsigned)((bbase + bL) * Sdim + s)) * 64u + (unsigned)(t0 + j);
      const float u = u01f(hash32(id));
      Lns[j] = __logf(u) - __logf(1.f - u);
    }
    lds_barrier();

    // ---- 2) gates = [x|h](16x128) @ Wcat^T(128x256) + bias, bf16 WMMA ----
    v16bf Areg[4];
#pragma unroll
    for (int kc = 0; kc < 4; ++kc) Areg[kc] = load_a_frag(Abf, col, kc, half);
#pragma unroll
    for (int tt = 0; tt < 2; ++tt) {
      v8f acc;
#pragma unroll
      for (int j = 0; j < 8; ++j) acc[j] = biasg[tt];   // bias per column N
#pragma unroll
      for (int kc = 0; kc < 4; ++kc)
        acc = __builtin_amdgcn_wmma_f32_16x16x32_bf16(
            false, Areg[kc], false, Bg[tt][kc], (short)0, acc, false, false);
      const int gcol = (w * 2 + tt) * 16 + col;
#pragma unroll
      for (int j = 0; j < 8; ++j) gatesS[(half * 8 + j) * 256 + gcol] = acc[j];
    }
    lds_barrier();

    // ---- 3) LSTM pointwise update; h -> bf16 into A cols 64..127 ----
    {
      const float4 gi = *(const float4*)&gatesS[bL * 256 +   0 + t0];
      const float4 gf = *(const float4*)&gatesS[bL * 256 +  64 + t0];
      const float4 gg = *(const float4*)&gatesS[bL * 256 + 128 + t0];
      const float4 go = *(const float4*)&gatesS[bL * 256 + 192 + t0];
      const float ia[4] = {gi.x, gi.y, gi.z, gi.w};
      const float fa[4] = {gf.x, gf.y, gf.z, gf.w};
      const float ga[4] = {gg.x, gg.y, gg.z, gg.w};
      const float oa[4] = {go.x, go.y, go.z, go.w};
#pragma unroll
      for (int j = 0; j < 4; ++j) {
        const float c = sigmoid_fast(fa[j]) * creg[j] + sigmoid_fast(ia[j]) * tanh_fast(ga[j]);
        creg[j] = c;
        Abf[bL * 128 + 64 + t0 + j] = (__bf16)(sigmoid_fast(oa[j]) * tanh_fast(c));
      }
    }
    lds_barrier();

    // ---- 4) p = sigmoid(h @ W_lin^T + b_lin), WMMA on waves 0..3 ----
    if (w < 4) {
      v8f acc;
#pragma unroll
      for (int j = 0; j < 8; ++j) acc[j] = biasl;
#pragma unroll
      for (int kc = 0; kc < 2; ++kc) {
        const v16bf a = load_a_frag(Abf, col, kc + 2, half);  // h lives in cols 64..127
        acc = __builtin_amdgcn_wmma_f32_16x16x32_bf16(
            false, a, false, Bl[kc], (short)0, acc, false, false);
      }
#pragma unroll
      for (int j = 0; j < 8; ++j)
        pS[(half * 8 + j) * 64 + (w * 16 + col)] = sigmoid_fast(acc[j]);
    }
    lds_barrier();

    // ---- 5) gating + attention partial reductions ----
    const int n = s >> 4, f = s & 15;
    const float u1n = U1s[n], u3f = U3s[f];
    const float4 p4 = *(const float4*)&pS[bL * 64 + t0];
    const float pa[4] = {p4.x, p4.y, p4.z, p4.w};
    const float xa[4] = {xr.x, xr.y, xr.z, xr.w};
    float4* lp = (float4*)&lAacc[f * 1024 + bL * 64 + t0];
    float4 lv = *lp;
    float la[4] = {lv.x, lv.y, lv.z, lv.w};
#pragma unroll
    for (int j = 0; j < 4; ++j) {
      // softmax([(1-p)+G0, p+G1])[1] = sigmoid(2p-1+(G1-G0))
      const float gt = sigmoid_fast(2.f * pa[j] - 1.f + Lns[j]);
      const float xm = xa[j] * gt;
      float ra = (f == 0 ? 0.f : rhs_acc[j]) + u3f * xm;   // rhs[b,n,t] = sum_f U3[f]*xm
      rhs_acc[j] = ra;
      if (f == 15) ws_rhs[((size_t)(bbase + bL) * Ndim + n) * Tdim + t0 + j] = ra;
      la[j] += u1n * xm;                                    // lhsA[b,t,f] += U1[n]*xm
    }
    *lp = make_float4(la[0], la[1], la[2], la[3]);

    xr = xr_next;
  }

  __syncthreads();
  // lAacc is [f][b*64+t]; ws_lhsA wants (b,t,f)
  for (int i = tid; i < 16 * 1024; i += 256) {
    const int f = i & 15, bt = i >> 4;
    ws_lhsA[(size_t)bbase * 64 * 16 + i] = lAacc[f * 1024 + bt];
  }
}

// ---------------------------------------------------------------------------
// Phase 2: attention epilogue per batch. Reassociated:
//   product[t,s] = sum_f lhsA[t,f] * ( sum_n U2[f,n]*rhs[n,s] )
//   E = Ve @ tanh(product + be); softmax over t.
// ---------------------------------------------------------------------------
__global__ __launch_bounds__(256, 1) void attn_kernel(
    const float* __restrict__ ws_rhs,
    const float* __restrict__ ws_lhsA,
    const float* __restrict__ U2,
    const float* __restrict__ be,
    const float* __restrict__ Ve,
    float* __restrict__ out)
{
  __shared__ float U2s[16 * 512];  // 32 KB
  __shared__ float lA[64 * 16];
  __shared__ float R2[16 * 64];
  __shared__ float TP[64 * 64];
  __shared__ float Es[64 * 64];
  __shared__ float mx[64];
  __shared__ float sm[64];

  const int b = blockIdx.x;
  const int tid = threadIdx.x;

  for (int i = tid; i < 16 * 512; i += 256) U2s[i] = U2[i];
  for (int i = tid; i < 64 * 16; i += 256) lA[i] = ws_lhsA[(size_t)b * 64 * 16 + i];
  __syncthreads();

  // R2[f][s] = sum_n U2[f][n] * rhs[b][n][s]
  for (int i = tid; i < 16 * 64; i += 256) {
    const int f = i >> 6, s2 = i & 63;
    float acc = 0.f;
    const float* rp = ws_rhs + ((size_t)b * Ndim) * Tdim + s2;
    for (int n = 0; n < Ndim; ++n) acc += U2s[f * Ndim + n] * rp[(size_t)n * Tdim];
    R2[i] = acc;
  }
  __syncthreads();

  // TP[t][s] = tanh( sum_f lA[t][f]*R2[f][s] + be[t][s] )
  for (int i = tid; i < 64 * 64; i += 256) {
    const int t = i >> 6, s2 = i & 63;
    float acc = be[i];
#pragma unroll
    for (int f = 0; f < 16; ++f) acc += lA[t * 16 + f] * R2[f * 64 + s2];
    TP[i] = tanh_fast(acc);
  }
  __syncthreads();

  // E[t][s] = sum_r Ve[t][r] * TP[r][s]
  for (int i = tid; i < 64 * 64; i += 256) {
    const int t = i >> 6, s2 = i & 63;
    float acc = 0.f;
    for (int r = 0; r < 64; ++r) acc += Ve[t * 64 + r] * TP[r * 64 + s2];
    Es[i] = acc;
  }
  __syncthreads();

  if (tid < 64) {
    float m = -3.4e38f;
    for (int t = 0; t < 64; ++t) m = fmaxf(m, Es[t * 64 + tid]);
    float sum = 0.f;
    for (int t = 0; t < 64; ++t) sum += __expf(Es[t * 64 + tid] - m);
    mx[tid] = m; sm[tid] = sum;
  }
  __syncthreads();

  for (int i = tid; i < 64 * 64; i += 256) {
    const int s2 = i & 63;
    out[(size_t)b * 4096 + i] = __expf(Es[i] - mx[s2]) / sm[s2];
  }
}

extern "C" void kernel_launch(void* const* d_in, const int* in_sizes, int n_in,
                              void* d_out, int out_size, void* d_ws, size_t ws_size,
                              hipStream_t stream)
{
  (void)in_sizes; (void)n_in; (void)out_size; (void)ws_size;
  const float* x     = (const float*)d_in[0];
  const float* U1    = (const float*)d_in[1];
  const float* U2    = (const float*)d_in[2];
  const float* U3    = (const float*)d_in[3];
  const float* be    = (const float*)d_in[4];
  const float* Ve    = (const float*)d_in[5];
  const float* W_ih  = (const float*)d_in[6];
  const float* W_hh  = (const float*)d_in[7];
  const float* b_ih  = (const float*)d_in[8];
  const float* b_hh  = (const float*)d_in[9];
  const float* W_lin = (const float*)d_in[10];
  const float* b_lin = (const float*)d_in[11];
  float* out = (float*)d_out;

  float* ws_rhs  = (float*)d_ws;                           // B*N*T floats (4 MB)
  float* ws_lhsA = ws_rhs + (size_t)Bdim * Ndim * Tdim;    // B*T*F floats

  lstm_gate_kernel<<<2, 256, 0, stream>>>(x, U1, U3, W_ih, W_hh, b_ih, b_hh,
                                          W_lin, b_lin, ws_rhs, ws_lhsA);
  attn_kernel<<<Bdim, 256, 0, stream>>>(ws_rhs, ws_lhsA, U2, be, Ve, out);
}
